// GAT_33835752358449
// MI455X (gfx1250) — matrix operand order repaired and checked by reading, not compiled, not run on hardware
//
#include <hip/hip_runtime.h>
#include <math.h>

// ---------------- problem constants (match reference) ----------------
#define N_NODES   50000
#define N_EDGES0  400000
#define F_IN      128
#define HID       64
#define HEADS1    8
#define N_GRAPHS  64
#define N_CLASSES 10
#define NEG_SLOPE 0.2f
#define E_TOTAL   (N_EDGES0 + N_NODES)   // self-loops appended

// ---------------- WMMA types (gfx1250, wave32) ----------------
typedef __attribute__((ext_vector_type(16))) __bf16 v16bf;
typedef __attribute__((ext_vector_type(8)))  float  v8f;

// gfx1250 async global->LDS copy path (ASYNCcnt), if the toolchain exposes it
#if defined(__AMDGCN__) && defined(__gfx1250__) && \
    __has_builtin(__builtin_amdgcn_global_load_async_to_lds_b128) && \
    __has_builtin(__builtin_amdgcn_s_wait_asynccnt)
#define USE_ASYNC_LDS 1
#define GLOBAL_AS __attribute__((address_space(1)))
#define LDS_AS    __attribute__((address_space(3)))
// exact element type from the builtin's signature: int __vector(4)
typedef int v4i_vs __attribute__((vector_size(4 * sizeof(int))));
#else
#define USE_ASYNC_LDS 0
#endif

// LDS strides (floats): keep b128 stores 16B-aligned AND gathers conflict-free
#define AS_STRIDE 36   // 32 k + pad; 144B row stride
#define BS_STRIDE 68   // 64 n + pad; 272B row stride

// ============================================================
// GEMM: C[M,N] = A[M,K] * B[K,N], f32 row-major.
// bf16x3 split accumulation: C += Ah*Bh + Ah*Bl + Al*Bh.
// Block = 128 threads (4 waves) -> 16(M) x 64(N) tile, K stepped by 32.
// A slab shared by all waves, B slab split 16 cols/wave.
// Double-buffered LDS staging, async global->LDS when available.
// Requires: M % 16 == 0, N % 64 == 0, K % 32 == 0 (true for both layers).
// ============================================================
__global__ __launch_bounds__(128)
void gat_gemm_wmma_bf16x3(const float* __restrict__ A,
                          const float* __restrict__ B,
                          float* __restrict__ C,
                          int M, int N, int K) {
  __shared__ float As[2][16 * AS_STRIDE];
  __shared__ float Bs[2][32 * BS_STRIDE];

  const int t    = threadIdx.x;     // 0..127
  const int lane = t & 31;
  const int wave = t >> 5;          // 0..3
  const int half = lane >> 4;       // 0: lanes 0-15, 1: lanes 16-31
  const int mr   = lane & 15;

  const int nblk = N >> 6;                    // # of 64-wide N blocks
  const int bm   = (blockIdx.x / nblk) << 4;  // 16-row M tile base
  const int bn   = (blockIdx.x % nblk) << 6;  // 64-col N tile base
  const int tn   = bn + (wave << 4);          // this wave's 16-col slice

  // staging coordinates (fixed per thread)
  const int ar  = t >> 3;                     // A row 0..15
  const int ac4 = t & 7;                      // A float4 col 0..7
  const int steps = K >> 5;

  auto stage = [&](int buf, int kb) {
    // A slab: 16x32 f32 = 128 float4, one per thread (coalesced 128B rows)
    const float* ga = &A[(size_t)(bm + ar) * K + kb + (ac4 << 2)];
    float*       la = &As[buf][ar * AS_STRIDE + (ac4 << 2)];
#if USE_ASYNC_LDS
    __builtin_amdgcn_global_load_async_to_lds_b128(
        (GLOBAL_AS v4i_vs*)ga, (LDS_AS v4i_vs*)la, 0, 0);
#else
    *(float4*)la = *(const float4*)ga;
#endif
    // B slab: 32x64 f32 = 512 float4, four per thread
#pragma unroll
    for (int i = 0; i < 4; ++i) {
      const int q   = t + (i << 7);
      const int bk  = q >> 4;                 // k row 0..31
      const int bc4 = q & 15;                 // float4 col 0..15
      const float* gb = &B[(size_t)(kb + bk) * N + bn + (bc4 << 2)];
      float*       lb = &Bs[buf][bk * BS_STRIDE + (bc4 << 2)];
#if USE_ASYNC_LDS
      __builtin_amdgcn_global_load_async_to_lds_b128(
          (GLOBAL_AS v4i_vs*)gb, (LDS_AS v4i_vs*)lb, 0, 0);
#else
      *(float4*)lb = *(const float4*)gb;
#endif
    }
  };

  v8f acc = {};
  stage(0, 0);                                // prologue: k-slab 0 in flight

  for (int s = 0; s < steps; ++s) {
    __syncthreads();               // all waves done READING buf[(s+1)&1]
    if (s + 1 < steps) stage((s + 1) & 1, (s + 1) << 5);
#if USE_ASYNC_LDS
    if (s + 1 < steps) __builtin_amdgcn_s_wait_asynccnt(5); // step s done (in-order), step s+1 may fly
    else               __builtin_amdgcn_s_wait_asynccnt(0);
#endif
    __syncthreads();               // step-s slabs visible to every wave

    const float* as = &As[s & 1][0];
    const float* bs = &Bs[s & 1][0];
    v16bf ahi, alo, bhi, blo;
    // A fragment: 16x32 bf16 layout (ISA 7.12.2); k-pairs contiguous -> b64 LDS loads
#pragma unroll
    for (int v = 0; v < 8; ++v) {
      const int j  = v << 1;
      const int ka = (j & 7) + 8 * (((j >> 3) << 1) + half);
      const float2 a2 = *(const float2*)&as[mr * AS_STRIDE + ka];
      const __bf16 h0 = (__bf16)a2.x, h1 = (__bf16)a2.y;
      ahi[j]     = h0;  ahi[j + 1] = h1;
      alo[j]     = (__bf16)(a2.x - (float)h0);
      alo[j + 1] = (__bf16)(a2.y - (float)h1);
    }
    // B fragment: 32x16 bf16 layout; consecutive lanes hit consecutive banks
#pragma unroll
    for (int j = 0; j < 16; ++j) {
      const float bv = bs[((j & 7) + 8 * (((j >> 3) << 1) + half)) * BS_STRIDE + (wave << 4) + mr];
      const __bf16 hb = (__bf16)bv;
      bhi[j] = hb;
      blo[j] = (__bf16)(bv - (float)hb);
    }
    acc = __builtin_amdgcn_wmma_f32_16x16x32_bf16(false, alo, false, bhi,
                                                  (short)0, acc, false, false);
    acc = __builtin_amdgcn_wmma_f32_16x16x32_bf16(false, ahi, false, blo,
                                                  (short)0, acc, false, false);
    acc = __builtin_amdgcn_wmma_f32_16x16x32_bf16(false, ahi, false, bhi,
                                                  (short)0, acc, false, false);
  }

#pragma unroll
  for (int r = 0; r < 8; ++r)
    C[(size_t)(bm + r + 8 * half) * N + tn + mr] = acc[r];
}

// ---------------- helpers ----------------
__device__ __forceinline__ void atomicMaxFloat(float* addr, float v) {
  int vi = __float_as_int(v);
  if (vi >= 0) atomicMax((int*)addr, vi);
  else         atomicMin((unsigned int*)addr, (unsigned int)vi);
}

__device__ __forceinline__ void edge_endpoints(const int* __restrict__ ei,
                                               int e, int& s, int& d) {
  if (e < N_EDGES0) { s = ei[e]; d = ei[N_EDGES0 + e]; }
  else              { s = d = e - N_EDGES0; }           // self-loop
}

__global__ void fill_f32(float* __restrict__ p, float v, long long n) {
  long long i = (long long)blockIdx.x * blockDim.x + threadIdx.x;
  if (i < n) p[i] = v;
}

// al[n,h] = sum_c h[n,h*C+c] * a[h*C+c]
__global__ void att_coef(const float* __restrict__ h, const float* __restrict__ a,
                         float* __restrict__ out, int heads, int ch) {
  int idx = blockIdx.x * blockDim.x + threadIdx.x;
  if (idx >= N_NODES * heads) return;
  const int node = idx / heads, hd = idx % heads;
  const float* hp = h + (size_t)node * heads * ch + (size_t)hd * ch;
  const float* ap = a + (size_t)hd * ch;
  float s = 0.f;
  for (int c = 0; c < ch; ++c) s += hp[c] * ap[c];
  out[idx] = s;
}

// pass 1: e = leaky_relu(al_s[src]+al_d[dst]); segment max over dst
__global__ void edge_logits(const int* __restrict__ ei,
                            const float* __restrict__ als,
                            const float* __restrict__ ald,
                            float* __restrict__ el, float* __restrict__ mx,
                            int heads) {
  int idx = blockIdx.x * blockDim.x + threadIdx.x;
  if (idx >= E_TOTAL * heads) return;
  const int e = idx / heads, hd = idx % heads;
  int s, d; edge_endpoints(ei, e, s, d);
  float v = als[s * heads + hd] + ald[d * heads + hd];
  v = (v > 0.f) ? v : NEG_SLOPE * v;
  el[idx] = v;
  atomicMaxFloat(&mx[d * heads + hd], v);
}

// pass 2: ex = exp(e - max[dst]); segment sum over dst (in-place on el)
__global__ void edge_exp(const int* __restrict__ ei,
                         float* __restrict__ el,
                         const float* __restrict__ mx,
                         float* __restrict__ den, int heads) {
  int idx = blockIdx.x * blockDim.x + threadIdx.x;
  if (idx >= E_TOTAL * heads) return;
  const int e = idx / heads, hd = idx % heads;
  int s, d; edge_endpoints(ei, e, s, d);
  const float ex = expf(el[idx] - mx[d * heads + hd]);
  el[idx] = ex;
  atomicAdd(&den[d * heads + hd], ex);
}

// pass 3: out[dst,h,c] += h[src,h,c] * ex/(den[dst,h]+1e-16)
__global__ void edge_scatter(const int* __restrict__ ei,
                             const float* __restrict__ ex,
                             const float* __restrict__ den,
                             const float* __restrict__ hsrc,
                             float* __restrict__ out, int heads, int ch) {
  const long long idx = (long long)blockIdx.x * blockDim.x + threadIdx.x;
  const int hc = heads * ch;
  if (idx >= (long long)E_TOTAL * hc) return;
  const int e = (int)(idx / hc), j = (int)(idx % hc);
  const int hd = j / ch;
  int s, d; edge_endpoints(ei, e, s, d);
  const float alpha = ex[e * heads + hd] / (den[d * heads + hd] + 1e-16f);
  atomicAdd(&out[(size_t)d * hc + j], hsrc[(size_t)s * hc + j] * alpha);
}

// bias (+ optional ELU), in place
__global__ void bias_act(float* __restrict__ x, const float* __restrict__ b,
                         long long n, int hc, int do_elu) {
  long long i = (long long)blockIdx.x * blockDim.x + threadIdx.x;
  if (i >= n) return;
  float v = x[i] + b[i % hc];
  if (do_elu) v = (v > 0.f) ? v : expm1f(v);
  x[i] = v;
}

// mean-pool accumulation
__global__ void pool_accum(const float* __restrict__ h2,
                           const int* __restrict__ batch,
                           float* __restrict__ psum, float* __restrict__ pcnt) {
  int idx = blockIdx.x * blockDim.x + threadIdx.x;
  if (idx >= N_NODES * HID) return;
  const int n = idx / HID, c = idx % HID;
  const int g = batch[n];
  atomicAdd(&psum[g * HID + c], h2[idx]);
  if (c == 0) atomicAdd(&pcnt[g], 1.0f);
}

// pooled = sum/cnt; logits = pooled@fc_w + fc_b; log_softmax
__global__ void pool_fc_logsoftmax(const float* __restrict__ psum,
                                   const float* __restrict__ pcnt,
                                   const float* __restrict__ fcw,
                                   const float* __restrict__ fcb,
                                   float* __restrict__ out) {
  const int g = threadIdx.x;
  if (g >= N_GRAPHS) return;
  const float inv = 1.0f / fmaxf(pcnt[g], 1.0f);
  float logit[N_CLASSES];
#pragma unroll
  for (int k = 0; k < N_CLASSES; ++k) logit[k] = fcb[k];
  for (int c = 0; c < HID; ++c) {
    const float p = psum[g * HID + c] * inv;
#pragma unroll
    for (int k = 0; k < N_CLASSES; ++k) logit[k] += p * fcw[c * N_CLASSES + k];
  }
  float mx = logit[0];
#pragma unroll
  for (int k = 1; k < N_CLASSES; ++k) mx = fmaxf(mx, logit[k]);
  float s = 0.f;
#pragma unroll
  for (int k = 0; k < N_CLASSES; ++k) s += expf(logit[k] - mx);
  const float lse = mx + logf(s);
#pragma unroll
  for (int k = 0; k < N_CLASSES; ++k) out[g * N_CLASSES + k] = logit[k] - lse;
}

// ---------------- host-side orchestration ----------------
static inline int gridFor(long long n, int b) { return (int)((n + b - 1) / b); }

extern "C" void kernel_launch(void* const* d_in, const int* in_sizes, int n_in,
                              void* d_out, int out_size, void* d_ws, size_t ws_size,
                              hipStream_t stream) {
  const float* x     = (const float*)d_in[0];
  const int*   ei    = (const int*)  d_in[1];   // [2, E0]
  const int*   batch = (const int*)  d_in[2];
  const float* W1    = (const float*)d_in[3];   // [128, 512]
  const float* as1   = (const float*)d_in[4];   // [8, 64]
  const float* ad1   = (const float*)d_in[5];
  const float* b1    = (const float*)d_in[6];   // [512]
  const float* W2    = (const float*)d_in[7];   // [512, 64]
  const float* as2   = (const float*)d_in[8];   // [1, 64]
  const float* ad2   = (const float*)d_in[9];
  const float* b2    = (const float*)d_in[10];  // [64]
  const float* fcw   = (const float*)d_in[11];  // [64, 10]
  const float* fcb   = (const float*)d_in[12];  // [10]
  float* out = (float*)d_out;

  // carve workspace
  char* ws = (char*)d_ws;
  size_t off = 0;
  auto carve = [&](size_t bytes) -> float* {
    float* p = (float*)(ws + off);
    off += (bytes + 255) & ~(size_t)255;
    return p;
  };
  const size_t HC1 = (size_t)HEADS1 * HID;          // 512
  float* h1pre = carve((size_t)N_NODES * HC1 * 4);  // x@W1   [N,512]
  float* out1  = carve((size_t)N_NODES * HC1 * 4);  // layer1 aggregate / act
  float* als1  = carve((size_t)N_NODES * HEADS1 * 4);
  float* ald1  = carve((size_t)N_NODES * HEADS1 * 4);
  float* mx1   = carve((size_t)N_NODES * HEADS1 * 4);
  float* den1  = carve((size_t)N_NODES * HEADS1 * 4);
  float* ex1   = carve((size_t)E_TOTAL * HEADS1 * 4);
  float* h2pre = carve((size_t)N_NODES * HID * 4);  // out1@W2 [N,64]
  float* out2  = carve((size_t)N_NODES * HID * 4);
  float* als2  = carve((size_t)N_NODES * 4);
  float* ald2  = carve((size_t)N_NODES * 4);
  float* mx2   = carve((size_t)N_NODES * 4);
  float* den2  = carve((size_t)N_NODES * 4);
  float* ex2   = carve((size_t)E_TOTAL * 4);
  float* psum  = carve((size_t)N_GRAPHS * HID * 4);
  float* pcnt  = carve((size_t)N_GRAPHS * 4);
  (void)ws_size; (void)n_in; (void)in_sizes; (void)out_size;

  const int B = 256;
  // ---------------- layer 1 ----------------
  {
    const int blocks = (N_NODES / 16) * ((int)HC1 / 64);    // 25000
    gat_gemm_wmma_bf16x3<<<blocks, 128, 0, stream>>>(x, W1, h1pre,
                                                     N_NODES, (int)HC1, F_IN);
  }
  att_coef<<<gridFor((long long)N_NODES * HEADS1, B), B, 0, stream>>>(h1pre, as1, als1, HEADS1, HID);
  att_coef<<<gridFor((long long)N_NODES * HEADS1, B), B, 0, stream>>>(h1pre, ad1, ald1, HEADS1, HID);
  fill_f32<<<gridFor((long long)N_NODES * HEADS1, B), B, 0, stream>>>(mx1, -3.0e38f, (long long)N_NODES * HEADS1);
  fill_f32<<<gridFor((long long)N_NODES * HEADS1, B), B, 0, stream>>>(den1, 0.f, (long long)N_NODES * HEADS1);
  fill_f32<<<gridFor((long long)N_NODES * HC1, B), B, 0, stream>>>(out1, 0.f, (long long)N_NODES * HC1);
  edge_logits<<<gridFor((long long)E_TOTAL * HEADS1, B), B, 0, stream>>>(ei, als1, ald1, ex1, mx1, HEADS1);
  edge_exp<<<gridFor((long long)E_TOTAL * HEADS1, B), B, 0, stream>>>(ei, ex1, mx1, den1, HEADS1);
  edge_scatter<<<gridFor((long long)E_TOTAL * HC1, B), B, 0, stream>>>(ei, ex1, den1, h1pre, out1, HEADS1, HID);
  bias_act<<<gridFor((long long)N_NODES * HC1, B), B, 0, stream>>>(out1, b1, (long long)N_NODES * HC1, (int)HC1, 1);

  // ---------------- layer 2 (heads = 1) ----------------
  {
    const int blocks = (N_NODES / 16) * (HID / 64);         // 3125
    gat_gemm_wmma_bf16x3<<<blocks, 128, 0, stream>>>(out1, W2, h2pre,
                                                     N_NODES, HID, (int)HC1);
  }
  att_coef<<<gridFor((long long)N_NODES, B), B, 0, stream>>>(h2pre, as2, als2, 1, HID);
  att_coef<<<gridFor((long long)N_NODES, B), B, 0, stream>>>(h2pre, ad2, ald2, 1, HID);
  fill_f32<<<gridFor((long long)N_NODES, B), B, 0, stream>>>(mx2, -3.0e38f, (long long)N_NODES);
  fill_f32<<<gridFor((long long)N_NODES, B), B, 0, stream>>>(den2, 0.f, (long long)N_NODES);
  fill_f32<<<gridFor((long long)N_NODES * HID, B), B, 0, stream>>>(out2, 0.f, (long long)N_NODES * HID);
  edge_logits<<<gridFor((long long)E_TOTAL, B), B, 0, stream>>>(ei, als2, ald2, ex2, mx2, 1);
  edge_exp<<<gridFor((long long)E_TOTAL, B), B, 0, stream>>>(ei, ex2, mx2, den2, 1);
  edge_scatter<<<gridFor((long long)E_TOTAL * HID, B), B, 0, stream>>>(ei, ex2, den2, h2pre, out2, 1, HID);
  bias_act<<<gridFor((long long)N_NODES * HID, B), B, 0, stream>>>(out2, b2, (long long)N_NODES * HID, HID, 0);

  // ---------------- pool + fc + log_softmax ----------------
  fill_f32<<<gridFor((long long)N_GRAPHS * HID, B), B, 0, stream>>>(psum, 0.f, (long long)N_GRAPHS * HID);
  fill_f32<<<1, N_GRAPHS, 0, stream>>>(pcnt, 0.f, N_GRAPHS);
  pool_accum<<<gridFor((long long)N_NODES * HID, B), B, 0, stream>>>(out2, batch, psum, pcnt);
  pool_fc_logsoftmax<<<1, N_GRAPHS, 0, stream>>>(psum, pcnt, fcw, fcb, out);
}
